// AGLISTA_40553081209415
// MI455X (gfx1250) — compile-verified, matrix operand order checked by reading, not run on hardware
//
#include <hip/hip_runtime.h>
#include <hip/hip_bf16.h>

// ---------------------------------------------------------------------------
// AGLISTA on MI455X (gfx1250): 16 fused LISTA iterations per 16-row tile.
//   GEMMs via v_wmma_f32_16x16x32_bf16 (f32 accumulate), 4 output tiles
//   blocked per k-loop so one LDS A-fragment feeds 4 WMMAs.
//   A pre-packed to WMMA B-fragment order in d_ws (bf16, 2MB + 2MB),
//   streamed from L2 (A is 4MB, fully L2-resident on 192MB L2).
//   x / z / b live in LDS for the whole recurrence (~273KB dynamic LDS).
// ---------------------------------------------------------------------------

typedef __attribute__((ext_vector_type(16))) __bf16        v16bf;
typedef __attribute__((ext_vector_type(8)))  float         v8f;
typedef __attribute__((ext_vector_type(8)))  unsigned int  v8u;

#define BB      8192
#define MM      512
#define NN      2048
#define KIT     16
#define PTOP    50

#define ROWS    16          // batch rows per workgroup
#define XS_STR  2052        // f32 stride (2048 + 4 pad) -> bank-conflict-free
#define GX_STR  2056        // bf16 stride (2048 + 8 pad)
#define BS_STR  520         // bf16 stride (512 + 8 pad)

// union region: gxs (bf16 16x2056 = 65792B) then reused as zs (f32 16x2052 = 131328B)
#define SMEM_BYTES (ROWS*XS_STR*4 /*xs*/ + ROWS*XS_STR*4 /*union*/ + ROWS*BS_STR*2 /*b*/ + 64 /*thr*/)

static __device__ __forceinline__ unsigned short f32_to_bf16_rne(float f) {
    unsigned int u = __builtin_bit_cast(unsigned int, f);
    unsigned int r = u + 0x7FFFu + ((u >> 16) & 1u);   // round-to-nearest-even
    return (unsigned short)(r >> 16);
}

static __device__ __forceinline__ v16bf make_frag(uint4 a, uint4 b) {
    v8u t;
    t[0] = a.x; t[1] = a.y; t[2] = a.z; t[3] = a.w;
    t[4] = b.x; t[5] = b.y; t[6] = b.z; t[7] = b.w;
    return __builtin_bit_cast(v16bf, t);
}

// ---------------------------------------------------------------------------
// Pack A (512x2048 f32, row-major) into WMMA B-fragment order, bf16.
// GEMM1 (b = gx @ A^T): B[k=n][col=m];  fragment (nk, mt): 32 lanes x 16 halves.
// ---------------------------------------------------------------------------
__global__ void aglista_pack1(const float* __restrict__ A, unsigned short* __restrict__ bp) {
    unsigned e = blockIdx.x * 256u + threadIdx.x;          // 64*32*32*16 = 1M
    if (e >= 64u * 32u * 32u * 16u) return;
    unsigned h    = e & 15u;
    unsigned lane = (e >> 4) & 31u;
    unsigned mt   = (e >> 9) & 31u;
    unsigned nk   = e >> 14;
    unsigned m = mt * 16u + (lane & 15u);
    unsigned k = ((h >> 3) << 4) + ((lane >> 4) << 3) + (h & 7u);
    unsigned n = nk * 32u + k;
    bp[e] = f32_to_bf16_rne(A[m * NN + n]);
}

// GEMM2 (c = b @ A): B[k=m][col=n];  fragment (mk, nt): 32 lanes x 16 halves.
__global__ void aglista_pack2(const float* __restrict__ A, unsigned short* __restrict__ bp) {
    unsigned e = blockIdx.x * 256u + threadIdx.x;          // 16*128*32*16 = 1M
    if (e >= 16u * 128u * 32u * 16u) return;
    unsigned h    = e & 15u;
    unsigned lane = (e >> 4) & 31u;
    unsigned nt   = (e >> 9) & 127u;
    unsigned mk   = e >> 16;
    unsigned n = nt * 16u + (lane & 15u);
    unsigned k = ((h >> 3) << 4) + ((lane >> 4) << 3) + (h & 7u);
    unsigned m = mk * 32u + k;
    bp[e] = f32_to_bf16_rne(A[m * NN + n]);
}

// ---------------------------------------------------------------------------
// Main fused kernel: one block = 16 batch rows, all 16 iterations on-chip.
// ---------------------------------------------------------------------------
__global__ void __launch_bounds__(256, 1)
aglista_main(const float* __restrict__ y,
             const float* __restrict__ gamma_p,
             const float* __restrict__ theta_p,
             const float* __restrict__ a_p,
             const float* __restrict__ v_p,
             const float* __restrict__ vu_p,
             const float* __restrict__ theta_init_p,
             const unsigned short* __restrict__ bpack1,
             const unsigned short* __restrict__ bpack2,
             float* __restrict__ out) {
    extern __shared__ char smem[];
    float*          xs  = (float*)smem;                                  // ROWS x XS_STR
    char*           un  = smem + ROWS * XS_STR * 4;
    unsigned short* gxs = (unsigned short*)un;                           // ROWS x GX_STR (phase 1)
    float*          zs  = (float*)un;                                    // ROWS x XS_STR (phase 2+)
    unsigned short* bsm = (unsigned short*)(un + ROWS * XS_STR * 4);     // ROWS x BS_STR
    unsigned int*   thr = (unsigned int*)((char*)bsm + ROWS * BS_STR * 2);

    const int tid  = threadIdx.x;
    const int wv   = tid >> 5;
    const int lane = tid & 31;
    const int lr   = lane & 15;    // A-frag row / C-frag col
    const int lh   = lane >> 4;    // lane half
    const int row0 = blockIdx.x * ROWS;

    // x := 0 (pads included)
    for (int s = tid; s < ROWS * XS_STR; s += 256) xs[s] = 0.f;
    __syncthreads();

    for (int it = 0; it < KIT; ++it) {
        const float tcur = (it == 0) ? theta_init_p[0] : theta_p[it];
        const float vui  = vu_p[it];
        const float vi   = v_p[it];
        const float gi   = gamma_p[it];
        const float thi  = theta_p[it];
        const float ai   = a_p[it];

        // ---- gain phase: gxs = bf16( (1 + t*vu*exp(-v*|x|)) * x ) ----
        for (int s = tid; s < ROWS * NN; s += 256) {
            int r = s >> 11, n = s & (NN - 1);
            float x = xs[r * XS_STR + n];
            float g = 1.f + tcur * vui * __expf(-vi * fabsf(x));
            gxs[r * GX_STR + n] = f32_to_bf16_rne(g * x);
        }
        __syncthreads();

        // ---- GEMM1: b(16x512) = gx(16x2048) @ A^T - y ----
        // 4 m-tiles per wave, blocked into ONE k-loop: 1 A-frag -> 4 WMMAs.
        {
            const int mtb = wv * 4;                          // tiles mtb..mtb+3
            v8f acc0 = {}, acc1 = {}, acc2 = {}, acc3 = {};
            const unsigned short* bbase =
                bpack1 + ((((unsigned)mtb) * 32u + lane) << 4);
            for (int nk = 0; nk < 64; ++nk) {
                const int k0 = nk * 32 + lh * 8;
                uint4 a0 = *(const uint4*)(gxs + lr * GX_STR + k0);
                uint4 a1 = *(const uint4*)(gxs + lr * GX_STR + k0 + 16);
                v16bf afrag = make_frag(a0, a1);
                const unsigned short* bp = bbase + ((unsigned)nk << 14);
                v16bf b0 = *(const v16bf*)(bp);              // tile stride = 512 halves
                v16bf b1 = *(const v16bf*)(bp + 512);
                v16bf b2 = *(const v16bf*)(bp + 1024);
                v16bf b3 = *(const v16bf*)(bp + 1536);
                if (nk + 1 < 64)
                    __builtin_prefetch(bbase + (((unsigned)nk + 1) << 14), 0, 3);
                acc0 = __builtin_amdgcn_wmma_f32_16x16x32_bf16(
                           false, afrag, false, b0, (short)0, acc0, false, false);
                acc1 = __builtin_amdgcn_wmma_f32_16x16x32_bf16(
                           false, afrag, false, b1, (short)0, acc1, false, false);
                acc2 = __builtin_amdgcn_wmma_f32_16x16x32_bf16(
                           false, afrag, false, b2, (short)0, acc2, false, false);
                acc3 = __builtin_amdgcn_wmma_f32_16x16x32_bf16(
                           false, afrag, false, b3, (short)0, acc3, false, false);
            }
            #pragma unroll
            for (int u = 0; u < 4; ++u) {                    // C/D layout: r=j+8*lh, m=mt*16+lr
                v8f acc = (u == 0) ? acc0 : (u == 1) ? acc1 : (u == 2) ? acc2 : acc3;
                const int mt = mtb + u;
                #pragma unroll
                for (int j = 0; j < 8; ++j) {
                    int r = j + lh * 8;
                    int m = mt * 16 + lr;
                    float val = acc[j] - y[(size_t)(row0 + r) * MM + m];
                    bsm[r * BS_STR + m] = f32_to_bf16_rne(val);
                }
            }
        }
        __syncthreads();   // b complete; gxs dead -> region becomes zs

        // ---- GEMM2: z(16x2048) = x - gamma * ( b(16x512) @ A ) ----
        // 16 n-tiles per wave = 4 passes x 4-tile blocks.
        for (int pass = 0; pass < 4; ++pass) {
            const int ntb = wv * 4 + pass * 32;              // tiles ntb..ntb+3
            v8f acc0 = {}, acc1 = {}, acc2 = {}, acc3 = {};
            const unsigned short* bbase =
                bpack2 + ((((unsigned)ntb) * 32u + lane) << 4);
            for (int mk = 0; mk < 16; ++mk) {
                const int k0 = mk * 32 + lh * 8;
                uint4 a0 = *(const uint4*)(bsm + lr * BS_STR + k0);
                uint4 a1 = *(const uint4*)(bsm + lr * BS_STR + k0 + 16);
                v16bf afrag = make_frag(a0, a1);
                const unsigned short* bp = bbase + ((unsigned)mk << 16);
                v16bf b0 = *(const v16bf*)(bp);
                v16bf b1 = *(const v16bf*)(bp + 512);
                v16bf b2 = *(const v16bf*)(bp + 1024);
                v16bf b3 = *(const v16bf*)(bp + 1536);
                if (mk + 1 < 16)
                    __builtin_prefetch(bbase + (((unsigned)mk + 1) << 16), 0, 3);
                acc0 = __builtin_amdgcn_wmma_f32_16x16x32_bf16(
                           false, afrag, false, b0, (short)0, acc0, false, false);
                acc1 = __builtin_amdgcn_wmma_f32_16x16x32_bf16(
                           false, afrag, false, b1, (short)0, acc1, false, false);
                acc2 = __builtin_amdgcn_wmma_f32_16x16x32_bf16(
                           false, afrag, false, b2, (short)0, acc2, false, false);
                acc3 = __builtin_amdgcn_wmma_f32_16x16x32_bf16(
                           false, afrag, false, b3, (short)0, acc3, false, false);
            }
            #pragma unroll
            for (int u = 0; u < 4; ++u) {
                v8f acc = (u == 0) ? acc0 : (u == 1) ? acc1 : (u == 2) ? acc2 : acc3;
                const int nt = ntb + u;
                #pragma unroll
                for (int j = 0; j < 8; ++j) {
                    int r = j + lh * 8;
                    int n = nt * 16 + lr;
                    float z = xs[r * XS_STR + n] - gi * acc[j];
                    zs[r * XS_STR + n] = z;
                }
            }
        }
        __syncthreads();

        // ---- top-50 per row: binary search over |z| bit-space (monotone) ----
        for (int rr = 0; rr < 2; ++rr) {
            const int r = wv + rr * 8;
            const uint4* zp = (const uint4*)(zs + r * XS_STR);
            uint4 regs[16];
            #pragma unroll
            for (int j = 0; j < 16; ++j) {                   // n = j*128 + lane*4 + (0..3)
                uint4 q = zp[j * 32 + lane];
                q.x &= 0x7FFFFFFFu; q.y &= 0x7FFFFFFFu;
                q.z &= 0x7FFFFFFFu; q.w &= 0x7FFFFFFFu;
                regs[j] = q;
            }
            unsigned lo = 0u, hi = 0x7F800000u;              // smallest u with count(>u) < P
            while (lo < hi) {
                unsigned mid = lo + ((hi - lo) >> 1);
                int c = 0;
                #pragma unroll
                for (int j = 0; j < 16; ++j) {
                    c += (regs[j].x > mid); c += (regs[j].y > mid);
                    c += (regs[j].z > mid); c += (regs[j].w > mid);
                }
                #pragma unroll
                for (int off = 16; off > 0; off >>= 1) c += __shfl_xor(c, off, 32);
                if (c >= PTOP) lo = mid + 1; else hi = mid;
            }
            if (lane == 0) thr[r] = lo;                      // keep == (|z| > thr) incl. ties
        }
        __syncthreads();

        // ---- shrink + overshoot, write back x ----
        for (int s = tid; s < ROWS * NN; s += 256) {
            int r = s >> 11, n = s & (NN - 1);
            int o = r * XS_STR + n;
            float z = zs[o];
            float x = xs[o];
            unsigned az = __builtin_bit_cast(unsigned, z) & 0x7FFFFFFFu;
            float absz = __builtin_bit_cast(float, az);
            float shr  = copysignf(fmaxf(absz - thi, 0.f), z);
            float xn   = (az > thr[r]) ? z : shr;
            float ov   = 1.f + ai / (fabsf(xn - x) + 0.01f);
            xs[o] = x + ov * (xn - x);
        }
        __syncthreads();
    }

    // ---- store x; block 0 zeroes the two (K,1) aux outputs ----
    for (int s = tid; s < ROWS * NN; s += 256) {
        int r = s >> 11, n = s & (NN - 1);
        out[(size_t)(row0 + r) * NN + n] = xs[r * XS_STR + n];
    }
    if (blockIdx.x == 0 && tid < 2 * KIT)
        out[(size_t)BB * NN + tid] = 0.f;
}

// ---------------------------------------------------------------------------
extern "C" void kernel_launch(void* const* d_in, const int* in_sizes, int n_in,
                              void* d_out, int out_size, void* d_ws, size_t ws_size,
                              hipStream_t stream) {
    (void)in_sizes; (void)n_in; (void)out_size; (void)ws_size;
    const float* y          = (const float*)d_in[0];
    const float* A          = (const float*)d_in[1];
    const float* gamma_p    = (const float*)d_in[2];
    const float* theta_p    = (const float*)d_in[3];
    const float* a_p        = (const float*)d_in[4];
    const float* v_p        = (const float*)d_in[5];
    const float* vu_p       = (const float*)d_in[6];
    const float* theta_init = (const float*)d_in[7];
    // d_in[8] = info (unused)

    unsigned short* bp1 = (unsigned short*)d_ws;                 // 2MB
    unsigned short* bp2 = bp1 + (size_t)MM * NN;                 // 2MB

    aglista_pack1<<<(MM * NN + 255) / 256, 256, 0, stream>>>(A, bp1);
    aglista_pack2<<<(MM * NN + 255) / 256, 256, 0, stream>>>(A, bp2);

    aglista_main<<<BB / ROWS, 256, SMEM_BYTES, stream>>>(
        y, gamma_p, theta_p, a_p, v_p, vu_p, theta_init, bp1, bp2, (float*)d_out);
}
